// ObjectAttentionBlock2D_24859270709419
// MI455X (gfx1250) — matrix operand order repaired and checked by reading, not compiled, not run on hardware
//
#include <hip/hip_runtime.h>
#include <hip/hip_bf16.h>

// ---------------- problem constants ----------------
#define B_    8
#define INC   512
#define KEYC  256
#define NPIX  16384   // 128*128
#define KOBJ  19
#define KPAD  32
#define BN_EPS 1e-5f

typedef __attribute__((ext_vector_type(16))) _Float16 v16h;
typedef __attribute__((ext_vector_type(8)))  float    v8f;

union FragA { uint4 u[2]; v16h h; };
union Pack8 { uint4 u; _Float16 h[8]; };

// ---------------- CDNA5 async global->LDS (ASYNCcnt-tracked) ----------------
// ISA §10.2: flat LDS address low 32 bits == LDS byte offset, so truncating the
// generic pointer yields the VDST LDS-address operand for the async op.
__device__ __forceinline__ void async_copy_b128(void* lds_dst, const void* gsrc) {
    unsigned loff = (unsigned)(size_t)lds_dst;
    unsigned long long ga = (unsigned long long)(size_t)gsrc;
    asm volatile("global_load_async_to_lds_b128 %0, %1, off"
                 :: "v"(loff), "v"(ga) : "memory");
}
__device__ __forceinline__ void wait_async0() {
#if __has_builtin(__builtin_amdgcn_s_wait_asynccnt)
    __builtin_amdgcn_s_wait_asynccnt(0);
#else
    asm volatile("s_wait_asynccnt 0x0" ::: "memory");
#endif
}

// ---------------- workspace layout (bytes) ----------------
static constexpr size_t SZ_XT  = (size_t)B_ * NPIX * INC * 2;   // 128 MB (also reused for ctx)
static constexpr size_t SZ_QT  = (size_t)B_ * NPIX * KEYC * 2;  // 64 MB
static constexpr size_t OFF_XT = 0;
static constexpr size_t OFF_CTX = 0;                            // reuse XT region after q computed
static constexpr size_t OFF_Q1 = OFF_XT + SZ_XT;
static constexpr size_t OFF_Q  = OFF_Q1 + SZ_QT;
static constexpr size_t OFF_W1 = OFF_Q  + SZ_QT;                // 256*512 f16
static constexpr size_t OFF_W2 = OFF_W1 + (size_t)256*512*2;    // 256*256 f16
static constexpr size_t OFF_WU = OFF_W2 + (size_t)256*256*2;    // 512*256 f16
static constexpr size_t OFF_B1 = OFF_WU + (size_t)512*256*2;    // 256 f32
static constexpr size_t OFF_B2 = OFF_B1 + 256*4;
static constexpr size_t OFF_BU = OFF_B2 + 256*4;
static constexpr size_t OFF_T1 = OFF_BU + 512*4;                // B*19*256 f32
static constexpr size_t OFF_KT = OFF_T1 + (size_t)B_*KOBJ*KEYC*4; // B*32*256 f16
static constexpr size_t OFF_VT = OFF_KT + (size_t)B_*KPAD*KEYC*2; // B*256*32 f16

// ---------------- fold BN into weight + bias ----------------
__global__ void foldw_kernel(const float* __restrict__ W, const float* __restrict__ g,
                             const float* __restrict__ bt, const float* __restrict__ mu,
                             const float* __restrict__ var,
                             _Float16* __restrict__ Wh, float* __restrict__ bias,
                             int Cout, int Cin) {
    int idx = blockIdx.x * 256 + threadIdx.x;
    if (idx >= Cout * Cin) return;
    int o = idx / Cin, c = idx % Cin;
    float s = g[o] * rsqrtf(var[o] + BN_EPS);
    Wh[idx] = (_Float16)(W[idx] * s);
    if (c == 0) bias[o] = bt[o] - mu[o] * s;
}

// ---------------- x [B][C][P] f32 -> xT [B][P][C] f16 (LDS transpose) ----------------
__global__ __launch_bounds__(256) void convx_kernel(const float* __restrict__ x,
                                                    _Float16* __restrict__ xT) {
    __shared__ float tile[32][33];
    const int b  = blockIdx.z;
    const int p0 = blockIdx.x * 32;
    const int c0 = blockIdx.y * 32;
    const int lx = threadIdx.x & 31;
    const int ly = threadIdx.x >> 5;   // 0..7
#pragma unroll
    for (int i = 0; i < 4; ++i) {
        int c = c0 + ly + 8 * i;
        tile[ly + 8 * i][lx] = x[((size_t)(b * INC + c)) * NPIX + p0 + lx];
    }
    __syncthreads();
#pragma unroll
    for (int i = 0; i < 4; ++i) {
        int p = p0 + ly + 8 * i;
        xT[((size_t)(b * NPIX + p)) * INC + c0 + lx] = (_Float16)tile[lx][ly + 8 * i];
    }
}

// ---------------- proxy path stage 1: t1[b][j][o] = cbr(proxy, ow1) ----------------
__global__ __launch_bounds__(256) void proxy1_kernel(const float* __restrict__ proxy,
                                                     const float* __restrict__ ow1,
                                                     const float* __restrict__ g, const float* __restrict__ bt,
                                                     const float* __restrict__ mu, const float* __restrict__ var,
                                                     float* __restrict__ t1) {
    const int b = blockIdx.x / KOBJ, j = blockIdx.x % KOBJ;
    const int o = threadIdx.x;
    const float* wr = ow1 + (size_t)o * INC;
    float acc = 0.f;
    for (int c = 0; c < INC; ++c)
        acc += wr[c] * proxy[((size_t)(b * INC + c)) * KOBJ + j];
    float s = g[o] * rsqrtf(var[o] + BN_EPS);
    t1[((size_t)(b * KOBJ + j)) * KEYC + o] = fmaxf(acc * s + (bt[o] - mu[o] * s), 0.f);
}

// ---------------- proxy path stage 2: kT [b][kpad][c], vT [b][c][kpad], padded f16 ----------------
__global__ __launch_bounds__(256) void proxy2_kernel(const float* __restrict__ proxy,
                                                     const float* __restrict__ t1,
                                                     const float* __restrict__ ow2,
                                                     const float* __restrict__ g2, const float* __restrict__ b2,
                                                     const float* __restrict__ m2, const float* __restrict__ v2,
                                                     const float* __restrict__ dw,
                                                     const float* __restrict__ dg, const float* __restrict__ db,
                                                     const float* __restrict__ dm, const float* __restrict__ dv,
                                                     _Float16* __restrict__ kT, _Float16* __restrict__ vT) {
    const int b = blockIdx.x;
    const int o = threadIdx.x;
    float s2 = g2[o] * rsqrtf(v2[o] + BN_EPS), sh2 = b2[o] - m2[o] * s2;
    float sd = dg[o] * rsqrtf(dv[o] + BN_EPS), shd = db[o] - dm[o] * sd;
    _Float16 vrow[KPAD];
#pragma unroll
    for (int j = 0; j < KPAD; ++j) vrow[j] = (_Float16)0.f;
    const float* w2r = ow2 + (size_t)o * KEYC;
    const float* dwr = dw  + (size_t)o * INC;
    for (int j = 0; j < KOBJ; ++j) {
        const float* t1r = t1 + ((size_t)(b * KOBJ + j)) * KEYC;
        float a = 0.f;
        for (int c = 0; c < KEYC; ++c) a += w2r[c] * t1r[c];
        kT[((size_t)(b * KPAD + j)) * KEYC + o] = (_Float16)fmaxf(a * s2 + sh2, 0.f);
        float vv = 0.f;
        for (int c = 0; c < INC; ++c) vv += dwr[c] * proxy[((size_t)(b * INC + c)) * KOBJ + j];
        vrow[j] = (_Float16)fmaxf(vv * sd + shd, 0.f);
    }
    for (int j = KOBJ; j < KPAD; ++j)
        kT[((size_t)(b * KPAD + j)) * KEYC + o] = (_Float16)0.f;
#pragma unroll
    for (int j = 0; j < KPAD; j += 8) {
        Pack8 pk;
#pragma unroll
        for (int r = 0; r < 8; ++r) pk.h[r] = vrow[j + r];
        *(uint4*)&vT[((size_t)(b * KEYC + o)) * KPAD + j] = pk.u;
    }
}

// ---------------- WMMA GEMM: Y = relu(W' x + bias), 128x128x32 tiles ----------------
// Double-buffered LDS fed by async global->LDS copies (ASYNCcnt).
// A = Wh [M][KD] (f16), B-operand rows = SrcT [B][NPIX][KD] (pixel-major f16).
// !F32OUT: write OutT [B][NPIX][M] f16 (transposed for next stage's B-operand)
//  F32OUT: write OutF [B][M][NPIX] f32 (final output)
template <int KD, bool F32OUT>
__global__ __launch_bounds__(256, 2) void gemm_kernel(const _Float16* __restrict__ Wh,
                                                      const _Float16* __restrict__ SrcT,
                                                      const float* __restrict__ bias,
                                                      _Float16* __restrict__ OutT,
                                                      float* __restrict__ OutF, int M) {
    __shared__ __align__(16) _Float16 Alds[2][128][32];
    __shared__ __align__(16) _Float16 Blds[2][128][32];   // stored transposed: [n][k]
    const int b  = blockIdx.z;
    const int n0 = blockIdx.x * 128;
    const int m0 = blockIdx.y * 128;
    const int tid  = threadIdx.x;
    const int lane = tid & 31;
    const int wv   = tid >> 5;        // 0..7
    const int wm   = wv & 1;          // M half (64)
    const int wn   = wv >> 1;         // N quarter (32)
    const int lm   = lane & 15;
    const int hl   = lane >> 4;

    const _Float16* Sb = SrcT + (size_t)b * NPIX * KD;
    const int r  = tid >> 1;          // 0..127
    const int ch = (tid & 1) * 8;     // f16 chunk offset

    auto load_tiles = [&](int buf, int k0) {
        async_copy_b128(&Alds[buf][r][ch], &Wh[(size_t)(m0 + r) * KD + k0 + ch]);
        async_copy_b128(&Blds[buf][r][ch], &Sb[(size_t)(n0 + r) * KD + k0 + ch]);
    };

    v8f acc[4][2];
    const v8f vzero = {0.f, 0.f, 0.f, 0.f, 0.f, 0.f, 0.f, 0.f};
#pragma unroll
    for (int mt = 0; mt < 4; ++mt)
#pragma unroll
        for (int nt = 0; nt < 2; ++nt) acc[mt][nt] = vzero;

    load_tiles(0, 0);
    wait_async0();
    __syncthreads();
    int buf = 0;
    const int ksteps = KD / 32;
    for (int kk = 0; kk < ksteps; ++kk) {
        if (kk + 1 < ksteps) load_tiles(buf ^ 1, (kk + 1) * 32);
        if (kk + 2 < ksteps) {   // global_prefetch_b8 toward L2 for tile after next
            int kp = (kk + 2) * 32;
            __builtin_prefetch(&Wh[(size_t)(m0 + r) * KD + kp], 0, 1);
            __builtin_prefetch(&Sb[(size_t)(n0 + r) * KD + kp], 0, 1);
        }
        FragA af[4], bfr[2];
#pragma unroll
        for (int mt = 0; mt < 4; ++mt) {
            int row = wm * 64 + mt * 16 + lm;
            af[mt].u[0] = *(const uint4*)&Alds[buf][row][hl * 8];
            af[mt].u[1] = *(const uint4*)&Alds[buf][row][16 + hl * 8];
        }
#pragma unroll
        for (int nt = 0; nt < 2; ++nt) {
            int col = wn * 32 + nt * 16 + lm;
            bfr[nt].u[0] = *(const uint4*)&Blds[buf][col][hl * 8];
            bfr[nt].u[1] = *(const uint4*)&Blds[buf][col][16 + hl * 8];
        }
#pragma unroll
        for (int mt = 0; mt < 4; ++mt)
#pragma unroll
            for (int nt = 0; nt < 2; ++nt)
                acc[mt][nt] = __builtin_amdgcn_wmma_f32_16x16x32_f16(
                    false, af[mt].h, false, bfr[nt].h, (short)0, acc[mt][nt], false, false);
        wait_async0();           // next buffer's async writes landed in LDS
        __syncthreads();
        buf ^= 1;
    }

    if (!F32OUT) {
        _Float16* Ob = OutT + (size_t)b * NPIX * M;
#pragma unroll
        for (int mt = 0; mt < 4; ++mt)
#pragma unroll
            for (int nt = 0; nt < 2; ++nt) {
                int p = n0 + wn * 32 + nt * 16 + lm;
                int mb = m0 + wm * 64 + mt * 16 + hl * 8;
                Pack8 pk;
#pragma unroll
                for (int rr = 0; rr < 8; ++rr)
                    pk.h[rr] = (_Float16)fmaxf(acc[mt][nt][rr] + bias[mb + rr], 0.f);
                *(uint4*)&Ob[(size_t)p * M + mb] = pk.u;
            }
    } else {
        float* Ob = OutF + (size_t)b * M * NPIX;
#pragma unroll
        for (int mt = 0; mt < 4; ++mt)
#pragma unroll
            for (int nt = 0; nt < 2; ++nt) {
                int p = n0 + wn * 32 + nt * 16 + lm;
                int mb = m0 + wm * 64 + mt * 16 + hl * 8;
#pragma unroll
                for (int rr = 0; rr < 8; ++rr)
                    Ob[(size_t)(mb + rr) * NPIX + p] = fmaxf(acc[mt][nt][rr] + bias[mb + rr], 0.f);
            }
    }
}

// ---------------- fused attention: sim (WMMA) -> softmax (shfl) -> ctx (WMMA) ----------------
__global__ __launch_bounds__(256, 2) void attn_kernel(const _Float16* __restrict__ qT,
                                                      const _Float16* __restrict__ kT,
                                                      const _Float16* __restrict__ vT,
                                                      _Float16* __restrict__ ctxT) {
    __shared__ __align__(16) _Float16 ks[KPAD][KEYC];      // 16 KB  [n=obj][c]
    __shared__ __align__(16) _Float16 vs[KEYC][KPAD];      // 16 KB  [c][k=obj]
    __shared__ __align__(16) _Float16 attn_s[8][16][KPAD]; // 8 KB A-fragment staging
    __shared__ __align__(16) _Float16 ctx_ls[8][16][16];   // 4 KB ctx tile transpose
    const int b = blockIdx.y;
    const int tid = threadIdx.x;
    {   // async-stage k^T and v^T into LDS (4 b128 per thread per buffer)
        const _Float16* ksrc = kT + (size_t)b * KPAD * KEYC;
        const _Float16* vsrc = vT + (size_t)b * KEYC * KPAD;
        _Float16* kdst = &ks[0][0];
        _Float16* vdst = &vs[0][0];
#pragma unroll
        for (int i = 0; i < 4; ++i) {
            int e = (tid + i * 256) * 8;
            async_copy_b128(kdst + e, ksrc + e);
            async_copy_b128(vdst + e, vsrc + e);
        }
    }
    wait_async0();
    __syncthreads();

    const int lane = tid & 31, wv = tid >> 5;
    const int lm = lane & 15, hl = lane >> 4;
    const int p0 = blockIdx.x * 128 + wv * 16;
    const _Float16* qb = qT + ((size_t)b * NPIX + p0) * KEYC;

    v8f s0 = {0.f,0.f,0.f,0.f,0.f,0.f,0.f,0.f};
    v8f s1 = s0;
#pragma unroll
    for (int c0 = 0; c0 < KEYC; c0 += 32) {
        FragA af, b0, b1;
        const _Float16* qrow = qb + (size_t)lm * KEYC + c0;
        af.u[0] = *(const uint4*)&qrow[hl * 8];
        af.u[1] = *(const uint4*)&qrow[16 + hl * 8];
        b0.u[0] = *(const uint4*)&ks[lm][c0 + hl * 8];
        b0.u[1] = *(const uint4*)&ks[lm][c0 + 16 + hl * 8];
        b1.u[0] = *(const uint4*)&ks[16 + lm][c0 + hl * 8];
        b1.u[1] = *(const uint4*)&ks[16 + lm][c0 + 16 + hl * 8];
        s0 = __builtin_amdgcn_wmma_f32_16x16x32_f16(false, af.h, false, b0.h, (short)0, s0, false, false);
        s1 = __builtin_amdgcn_wmma_f32_16x16x32_f16(false, af.h, false, b1.h, (short)0, s1, false, false);
    }

    // softmax over 19 objects, per pixel row m = r + 8*hl (16-lane butterfly)
    const float sc = 0.0625f;  // 256^-0.5
    const bool ok1 = lm < (KOBJ - 16);
#pragma unroll
    for (int rr = 0; rr < 8; ++rr) {
        float x0 = s0[rr] * sc;
        float x1 = ok1 ? s1[rr] * sc : -1e30f;
        float mx = fmaxf(x0, x1);
#pragma unroll
        for (int d = 1; d < 16; d <<= 1) mx = fmaxf(mx, __shfl_xor(mx, d, 32));
        float e0 = __expf(x0 - mx);
        float e1 = ok1 ? __expf(x1 - mx) : 0.f;
        float sm = e0 + e1;
#pragma unroll
        for (int d = 1; d < 16; d <<= 1) sm += __shfl_xor(sm, d, 32);
        float inv = 1.0f / sm;
        int m = rr + 8 * hl;
        attn_s[wv][m][lm]      = (_Float16)(e0 * inv);
        attn_s[wv][m][16 + lm] = (_Float16)(e1 * inv);
    }
    __syncthreads();

    FragA at;
    at.u[0] = *(const uint4*)&attn_s[wv][lm][8 * hl];
    at.u[1] = *(const uint4*)&attn_s[wv][lm][16 + 8 * hl];
    _Float16* cb = ctxT + ((size_t)b * NPIX + p0) * KEYC;
#pragma unroll
    for (int nc = 0; nc < KEYC / 16; ++nc) {
        FragA bv;
        int chb = nc * 16 + lm;
        bv.u[0] = *(const uint4*)&vs[chb][hl * 8];
        bv.u[1] = *(const uint4*)&vs[chb][16 + hl * 8];
        v8f c = {0.f,0.f,0.f,0.f,0.f,0.f,0.f,0.f};
        c = __builtin_amdgcn_wmma_f32_16x16x32_f16(false, at.h, false, bv.h, (short)0, c, false, false);
        // transpose 16x16 tile through per-wave LDS (DS ops are in-order per wave),
        // then one coalesced b128 store per lane
#pragma unroll
        for (int rr = 0; rr < 8; ++rr)
            ctx_ls[wv][rr + 8 * hl][lm] = (_Float16)c[rr];
        uint4 pk = *(const uint4*)&ctx_ls[wv][lane >> 1][(lane & 1) * 8];
        *(uint4*)&cb[(size_t)(lane >> 1) * KEYC + nc * 16 + (lane & 1) * 8] = pk;
    }
}

// ---------------- host launcher ----------------
extern "C" void kernel_launch(void* const* d_in, const int* in_sizes, int n_in,
                              void* d_out, int out_size, void* d_ws, size_t ws_size,
                              hipStream_t stream) {
    (void)in_sizes; (void)n_in; (void)out_size; (void)ws_size;
    const float* x     = (const float*)d_in[0];
    const float* proxy = (const float*)d_in[1];
    const float* pw1 = (const float*)d_in[2];
    const float *pg1 = (const float*)d_in[3], *pb1 = (const float*)d_in[4],
                *pm1 = (const float*)d_in[5], *pv1 = (const float*)d_in[6];
    const float* pw2 = (const float*)d_in[7];
    const float *pg2 = (const float*)d_in[8], *pb2 = (const float*)d_in[9],
                *pm2 = (const float*)d_in[10], *pv2 = (const float*)d_in[11];
    const float* ow1 = (const float*)d_in[12];
    const float *og1 = (const float*)d_in[13], *ob1 = (const float*)d_in[14],
                *om1 = (const float*)d_in[15], *ov1 = (const float*)d_in[16];
    const float* ow2 = (const float*)d_in[17];
    const float *og2 = (const float*)d_in[18], *ob2 = (const float*)d_in[19],
                *om2 = (const float*)d_in[20], *ov2 = (const float*)d_in[21];
    const float* dw  = (const float*)d_in[22];
    const float *dg = (const float*)d_in[23], *db = (const float*)d_in[24],
                *dm = (const float*)d_in[25], *dv = (const float*)d_in[26];
    const float* uw  = (const float*)d_in[27];
    const float *ug = (const float*)d_in[28], *ub = (const float*)d_in[29],
                *um = (const float*)d_in[30], *uv = (const float*)d_in[31];

    char* ws = (char*)d_ws;
    _Float16* XT  = (_Float16*)(ws + OFF_XT);
    _Float16* CTX = (_Float16*)(ws + OFF_CTX);
    _Float16* Q1  = (_Float16*)(ws + OFF_Q1);
    _Float16* Q   = (_Float16*)(ws + OFF_Q);
    _Float16* W1h = (_Float16*)(ws + OFF_W1);
    _Float16* W2h = (_Float16*)(ws + OFF_W2);
    _Float16* WUh = (_Float16*)(ws + OFF_WU);
    float*    B1f = (float*)(ws + OFF_B1);
    float*    B2f = (float*)(ws + OFF_B2);
    float*    BUf = (float*)(ws + OFF_BU);
    float*    T1f = (float*)(ws + OFF_T1);
    _Float16* KTh = (_Float16*)(ws + OFF_KT);
    _Float16* VTh = (_Float16*)(ws + OFF_VT);

    // 1) fold BN into f16 weights
    foldw_kernel<<<(256 * 512 + 255) / 256, 256, 0, stream>>>(pw1, pg1, pb1, pm1, pv1, W1h, B1f, 256, 512);
    foldw_kernel<<<(256 * 256 + 255) / 256, 256, 0, stream>>>(pw2, pg2, pb2, pm2, pv2, W2h, B2f, 256, 256);
    foldw_kernel<<<(512 * 256 + 255) / 256, 256, 0, stream>>>(uw,  ug,  ub,  um,  uv,  WUh, BUf, 512, 256);

    // 2) x -> xT f16
    convx_kernel<<<dim3(NPIX / 32, INC / 32, B_), 256, 0, stream>>>(x, XT);

    // 3) proxy path -> kT, vT (f16, K padded to 32)
    proxy1_kernel<<<B_ * KOBJ, 256, 0, stream>>>(proxy, ow1, og1, ob1, om1, ov1, T1f);
    proxy2_kernel<<<B_, 256, 0, stream>>>(proxy, T1f, ow2, og2, ob2, om2, ov2,
                                          dw, dg, db, dm, dv, KTh, VTh);

    // 4) q path: two WMMA GEMMs
    gemm_kernel<512, false><<<dim3(NPIX / 128, 2, B_), 256, 0, stream>>>(W1h, XT, B1f, Q1, nullptr, 256);
    gemm_kernel<256, false><<<dim3(NPIX / 128, 2, B_), 256, 0, stream>>>(W2h, Q1, B2f, Q,  nullptr, 256);

    // 5) fused attention -> ctxT (reuses XT region; x no longer needed)
    attn_kernel<<<dim3(NPIX / 128, B_), 256, 0, stream>>>(Q, KTh, VTh, CTX);

    // 6) f_up: final WMMA GEMM, fp32 out with bias+ReLU
    gemm_kernel<256, true><<<dim3(NPIX / 128, 4, B_), 256, 0, stream>>>(WUh, CTX, BUf, nullptr, (float*)d_out, 512);
}
